// LinearAttention_72834055405759
// MI455X (gfx1250) — matrix operand order repaired and checked by reading, not compiled, not run on hardware
//
#include <hip/hip_runtime.h>

typedef __attribute__((ext_vector_type(16))) _Float16 v16h;
typedef __attribute__((ext_vector_type(8)))  float    v8f;

#define BB 2
#define HH 16
#define LL 2048
#define DD 64
#define BHN (BB*HH)
#define CTXN ((size_t)BHN*LL*DD)   // 4,194,304 elements

union FragU { unsigned int u[8]; v16h h; };

__device__ __forceinline__ float featf(float x) {
  // elu(x)+1 == x>0 ? x+1 : exp(x)
  return x > 0.f ? x + 1.f : __expf(x);
}

// ---------------- Kernel 1: featurize Q and K to f16 ----------------
__global__ void featurize_kernel(const float* __restrict__ q,
                                 const float* __restrict__ k,
                                 _Float16* __restrict__ qf,
                                 _Float16* __restrict__ kf) {
  size_t i = (size_t)blockIdx.x * blockDim.x + threadIdx.x;
  qf[i] = (_Float16)featf(q[i]);
  kf[i] = (_Float16)featf(k[i]);
}

// ---------------- Kernel 2: V [bh][l][d] -> Vt f16 [bh][d][l] ----------------
__global__ void transposeV_kernel(const float* __restrict__ v,
                                  _Float16* __restrict__ vt) {
  __shared__ _Float16 tile[64][66];
  const int bh = blockIdx.y;
  const int l0 = blockIdx.x * 64;
  #pragma unroll
  for (int i = 0; i < 16; i++) {
    int lin = threadIdx.x + i * 256;
    int ll = lin >> 6, d = lin & 63;
    tile[ll][d] = (_Float16)v[((size_t)bh * LL + l0 + ll) * DD + d];
  }
  __syncthreads();
  #pragma unroll
  for (int i = 0; i < 16; i++) {
    int lin = threadIdx.x + i * 256;
    int d = lin >> 6, ll = lin & 63;
    vt[((size_t)bh * DD + d) * LL + l0 + ll] = tile[ll][d];
  }
}

// ---------------- Kernel 3: softmax attention with WMMA ----------------
// Grid: (LL/128, BHN), block 256 (8 waves). Each wave owns 16 query rows.
__global__ __launch_bounds__(256, 1)
void attn_kernel(const _Float16* __restrict__ qf,
                 const _Float16* __restrict__ kf,
                 const _Float16* __restrict__ vt,
                 float* __restrict__ outCtx,
                 float* __restrict__ outAttn) {
  __shared__ _Float16 plds[8][16][36];   // per-wave 16x32 P staging (+pad)

  const int bh   = blockIdx.y;
  const int wave = threadIdx.x >> 5;
  const int lane = threadIdx.x & 31;
  const int l16  = lane & 15;
  const int hi   = lane >> 4;            // half-wave select
  const int q0   = blockIdx.x * 128 + wave * 16;

  // ---- Load Q A-fragments (16x32 f16 each, d=0..31 and d=32..63) ----
  // A layout: lane -> M=l16 ; VGPR v, half h -> K = 2v + h + 16*hi
  FragU aq0, aq1;
  {
    const _Float16* qp = qf + ((size_t)bh * LL + q0 + l16) * DD;
    #pragma unroll
    for (int v = 0; v < 8; v++) {
      int kl = 2 * v + (hi ? 16 : 0);
      aq0.u[v] = *(const unsigned int*)(qp + kl);
      aq1.u[v] = *(const unsigned int*)(qp + 32 + kl);
    }
  }

  float mr[8], sr[8];
  #pragma unroll
  for (int r = 0; r < 8; r++) { mr[r] = -1e30f; sr[r] = 0.f; }

  // ======== Pass A: online row-max and sum(exp) over all 2048 keys ========
  for (int kt = 0; kt < LL / 16; kt++) {
    // B layout (32x16): lane -> N=l16 ; VGPR v, half h -> K = 2v + h + 16*hi
    FragU bk0, bk1;
    const _Float16* kp = kf + ((size_t)bh * LL + kt * 16 + l16) * DD;
    #pragma unroll
    for (int v = 0; v < 8; v++) {
      int dl = 2 * v + (hi ? 16 : 0);
      bk0.u[v] = *(const unsigned int*)(kp + dl);
      bk1.u[v] = *(const unsigned int*)(kp + 32 + dl);
    }
    v8f c = {};
    c = __builtin_amdgcn_wmma_f32_16x16x32_f16(false, aq0.h, false, bk0.h, (short)0, c, false, false);
    c = __builtin_amdgcn_wmma_f32_16x16x32_f16(false, aq1.h, false, bk1.h, (short)0, c, false, false);
    #pragma unroll
    for (int r = 0; r < 8; r++) {
      float s  = c[r];
      float nm = fmaxf(mr[r], s);
      sr[r] = sr[r] * __expf(mr[r] - nm) + __expf(s - nm);
      mr[r] = nm;
    }
  }

  // ---- 16-lane reductions: exact row max and softmax denominator ----
  float rowM[8], rowInv[8];
  #pragma unroll
  for (int r = 0; r < 8; r++) {
    float M = mr[r];
    #pragma unroll
    for (int d = 1; d < 16; d <<= 1) M = fmaxf(M, __shfl_xor(M, d, 32));
    float S = sr[r] * __expf(mr[r] - M);
    #pragma unroll
    for (int d = 1; d < 16; d <<= 1) S += __shfl_xor(S, d, 32);
    rowM[r]   = M;
    rowInv[r] = 1.f / S;
  }

  v8f oc[4] = {v8f{}, v8f{}, v8f{}, v8f{}};

  // ======== Pass B: write normalized attn, accumulate O += P*V ========
  for (int kt = 0; kt < LL / 16; kt++) {
    FragU bk0, bk1;
    const _Float16* kp = kf + ((size_t)bh * LL + kt * 16 + l16) * DD;
    #pragma unroll
    for (int v = 0; v < 8; v++) {
      int dl = 2 * v + (hi ? 16 : 0);
      bk0.u[v] = *(const unsigned int*)(kp + dl);
      bk1.u[v] = *(const unsigned int*)(kp + 32 + dl);
    }
    v8f c = {};
    c = __builtin_amdgcn_wmma_f32_16x16x32_f16(false, aq0.h, false, bk0.h, (short)0, c, false, false);
    c = __builtin_amdgcn_wmma_f32_16x16x32_f16(false, aq1.h, false, bk1.h, (short)0, c, false, false);

    float p[8];
    #pragma unroll
    for (int r = 0; r < 8; r++) p[r] = __expf(c[r] - rowM[r]) * rowInv[r];

    // write attn tile: row = q0 + r + 8*hi, col = kt*16 + l16  (C layout)
    {
      float* ap = outAttn + ((size_t)bh * LL + q0 + (hi ? 8 : 0)) * LL + (size_t)kt * 16 + l16;
      #pragma unroll
      for (int r = 0; r < 8; r++) ap[(size_t)r * LL] = p[r];
    }

    // stage P (f16) into per-wave LDS tile for C->A re-layout
    const int colb = (kt & 1) * 16;
    #pragma unroll
    for (int r = 0; r < 8; r++)
      plds[wave][r + (hi ? 8 : 0)][colb + l16] = (_Float16)p[r];

    if (kt & 1) {
      // A-fragment of P (16 rows x 32 keys)
      FragU apf;
      #pragma unroll
      for (int v = 0; v < 8; v++) {
        int kl = 2 * v + (hi ? 16 : 0);
        apf.u[v] = *(const unsigned int*)&plds[wave][l16][kl];
      }
      const int kb = (kt - 1) * 16;   // 32-key base
      #pragma unroll
      for (int nb = 0; nb < 4; nb++) {
        // B-fragment of V (32 keys x 16 d): lane -> N = nb*16+l16 (d), K = 2v+h+16*hi (key)
        FragU bv;
        const _Float16* vq = vt + ((size_t)bh * DD + nb * 16 + l16) * LL + kb;
        #pragma unroll
        for (int v = 0; v < 8; v++) {
          int kk = 2 * v + (hi ? 16 : 0);
          bv.u[v] = *(const unsigned int*)(vq + kk);
        }
        oc[nb] = __builtin_amdgcn_wmma_f32_16x16x32_f16(false, apf.h, false, bv.h, (short)0, oc[nb], false, false);
      }
    }
  }

  // ---- store context: row = q0 + r + 8*hi, col = nb*16 + l16 ----
  float* op = outCtx + ((size_t)bh * LL + q0 + (hi ? 8 : 0)) * DD + l16;
  #pragma unroll
  for (int nb = 0; nb < 4; nb++) {
    #pragma unroll
    for (int r = 0; r < 8; r++)
      op[(size_t)r * DD + nb * 16] = oc[nb][r];
  }
}

extern "C" void kernel_launch(void* const* d_in, const int* in_sizes, int n_in,
                              void* d_out, int out_size, void* d_ws, size_t ws_size,
                              hipStream_t stream) {
  const float* q = (const float*)d_in[0];
  const float* k = (const float*)d_in[1];
  const float* v = (const float*)d_in[2];

  _Float16* qf = (_Float16*)d_ws;   // CTXN f16
  _Float16* kf = qf + CTXN;         // CTXN f16
  _Float16* vt = kf + CTXN;         // CTXN f16 (transposed V)

  float* outCtx  = (float*)d_out;
  float* outAttn = outCtx + CTXN;

  featurize_kernel<<<(int)(CTXN / 256), 256, 0, stream>>>(q, k, qf, kf);
  transposeV_kernel<<<dim3(LL / 64, BHN), 256, 0, stream>>>(v, vt);
  attn_kernel<<<dim3(LL / 128, BHN), 256, 0, stream>>>(qf, kf, vt, outCtx, outAttn);
}
